// SingleLayerLSTM_71468255806047
// MI455X (gfx1250) — matrix operand order repaired and checked by reading, not compiled, not run on hardware
//
#include <hip/hip_runtime.h>
#include <hip/hip_bf16.h>

// CDNA5 wave32 WMMA vector types
typedef __bf16 bf16x16 __attribute__((ext_vector_type(16)));
typedef __bf16 bf16x8  __attribute__((ext_vector_type(8)));
typedef float  f32x8   __attribute__((ext_vector_type(8)));

#define DEVINL __device__ __forceinline__

namespace cfg {
constexpr int B = 128, T = 2048, D = 64, H = 256, O = 32;
constexpr int KTOT = D + H;       // 320 (u_t ++ h)
constexpr int NKG  = KTOT / 32;   // 10 K-chunks for gate GEMMs
constexpr int NKO  = H / 32;      // 8  K-chunks for output GEMM
constexpr int LDA  = KTOT + 8;    // 328 -> row pitch 656B = 41*16B (16B-aligned frags)
constexpr int THREADS = 512;      // 16 wave32 per workgroup
}

DEVINL float sigm(float x) { return 1.0f / (1.0f + __expf(-x)); }

DEVINL float ftanh(float x) {
#if __has_builtin(__builtin_amdgcn_tanhf)
  return __builtin_amdgcn_tanhf(x);
#elif __has_builtin(__builtin_amdgcn_tanh_f32)
  return __builtin_amdgcn_tanh_f32(x);
#else
  // stable tanh via v_exp_f32: saturates correctly for large |x|
  float e = __expf(2.0f * x);
  return 1.0f - 2.0f / (e + 1.0f);
#endif
}

// 16x32 bf16 A fragment from LDS row-major [16 x LDA] (wave32 layout, contiguous K):
// lane<16: M=lane, K = colBase + {0..15}; lane>=16: M=lane-16, K += 16.
DEVINL bf16x16 load_a_frag(const __bf16* sA, int lane, int colBase) {
  const __bf16* p = sA + (lane & 15) * cfg::LDA + colBase + 16 * (lane >> 4);
  bf16x8 lo = *(const bf16x8*)(p);       // K = base+0..7
  bf16x8 hi = *(const bf16x8*)(p + 8);   // K = base+8..15
  bf16x16 a;
#pragma unroll
  for (int j = 0; j < 8; ++j) { a[j] = lo[j]; a[8 + j] = hi[j]; }
  return a;
}

// Contiguous 32B B fragment (packed layout) -> two aligned 16B loads
DEVINL bf16x16 load_b_frag(const __bf16* p) {
  bf16x8 lo = *(const bf16x8*)(p);
  bf16x8 hi = *(const bf16x8*)(p + 8);
  bf16x16 b;
#pragma unroll
  for (int j = 0; j < 8; ++j) { b[j] = lo[j]; b[8 + j] = hi[j]; }
  return b;
}

// ---------------------------------------------------------------------------
// Pack fp32 row-major W[Ktot x Ntot] into bf16 wave32 B-fragment order:
// dst[ ((nt*nK + kci)*32 + lane)*16 + j ] = W[ kci*32 + 16*(lane>=16) + j ][ nt*16 + (lane&15) ]
// so a lane's B fragment is one contiguous 32B chunk.
// ---------------------------------------------------------------------------
__global__ void pack_weights_kernel(const float* __restrict__ src,
                                    __bf16* __restrict__ dst,
                                    int Ktot, int Ntot) {
  int nK = Ktot >> 5;
  long long total = (long long)(Ntot >> 4) * nK * 512;
  long long idx = (long long)blockIdx.x * blockDim.x + threadIdx.x;
  if (idx >= total) return;
  int hi   = (int)(idx & 15);
  int lane = (int)((idx >> 4) & 31);
  long long blk = idx >> 9;            // nt*nK + kci
  int kci = (int)(blk % nK);
  int nt  = (int)(blk / nK);
  int K = kci * 32 + ((lane >> 4) << 4) + hi;
  int N = nt * 16 + (lane & 15);
  dst[idx] = (__bf16)src[(long long)K * Ntot + N];
}

// ---------------------------------------------------------------------------
// Persistent LSTM scan: one workgroup per 16-row batch tile, full T loop.
// c lives in accumulator VGPRs; h (bf16) + W_r + W_out live in LDS;
// W_fiz streams from L2 every step (laundered offsets stop LICM/spilling).
// ---------------------------------------------------------------------------
__global__ __launch_bounds__(cfg::THREADS, 1)
void lstm_scan_kernel(const float* __restrict__ u,        // [B,T,D]
                      const float* __restrict__ x0,       // [B,1,2H]
                      const float* __restrict__ bias_fiz, // [3H]
                      const float* __restrict__ bias_r,   // [H]
                      const float* __restrict__ b_out,    // [O]
                      const __bf16* __restrict__ Wfiz,    // packed [320x768]
                      const __bf16* __restrict__ Wr,      // packed [320x256]
                      const __bf16* __restrict__ Wout,    // packed [256x32]
                      float* __restrict__ out)            // [B,T,O]
{
  using namespace cfg;
  __shared__ __bf16 sA[16 * LDA];      // 10.5 KB  A panel [16 x (u_t ++ h)]
  __shared__ __bf16 sWr[KTOT * H];     // 160 KB   packed W_r (whole matrix)
  __shared__ __bf16 sWout[H * O];      // 16 KB    packed W_out

  const int tid    = threadIdx.x;
  const int lane   = tid & 31;
  const int wv     = tid >> 5;        // wave 0..15
  const int b0     = blockIdx.x * 16; // batch base
  const int nloc   = lane & 15;       // column within 16-wide tile
  const int rhalf  = lane >> 4;       // 0/1 -> row halves of C layout
  const int colH   = wv * 16 + nloc;  // hidden column owned by this wave/lane

  // Biases (column-broadcast in the C layout)
  const float bF = bias_fiz[colH];
  const float bI = bias_fiz[H + colH];
  const float bZ = bias_fiz[2 * H + colH];
  const float bR = bias_r[colH];
  float bO = 0.0f;
  if (wv < 2) bO = b_out[wv * 16 + nloc];

  // c0 directly into C-matrix accumulator layout (element v -> row v+8*rhalf)
  f32x8 c = {};
#pragma unroll
  for (int v = 0; v < 8; ++v) {
    int m = v + 8 * rhalf;
    c[v] = x0[(long long)(b0 + m) * (2 * H) + H + colH];
  }

  // Cooperative preload: h0 -> LDS cols [D,D+H), u_{t=0} -> LDS cols [0,D)
  for (int i = tid; i < 16 * H; i += THREADS) {
    int m = i >> 8, n = i & 255;
    sA[m * LDA + D + n] = (__bf16)x0[(long long)(b0 + m) * (2 * H) + n];
  }
  for (int i = tid; i < 16 * D; i += THREADS) {
    int m = i >> 6, n = i & 63;
    sA[m * LDA + n] = (__bf16)u[((long long)(b0 + m) * T + 0) * D + n];
  }
  // Cooperative preload of packed W_r and W_out into LDS (16B vector copies)
  {
    const int4* src = (const int4*)Wr;
    int4* dst = (int4*)sWr;
    for (int i = tid; i < (KTOT * H) / 8; i += THREADS) dst[i] = src[i];
    const int4* srco = (const int4*)Wout;
    int4* dsto = (int4*)sWout;
    for (int i = tid; i < (H * O) / 8; i += THREADS) dsto[i] = srco[i];
  }

  // Per-wave packed weight bases (fiz n-tiles: f=wv, i=16+wv, z=32+wv)
  const __bf16* pF = Wfiz + (long long)(wv)      * NKG * 512;
  const __bf16* pI = Wfiz + (long long)(16 + wv) * NKG * 512;
  const __bf16* pZ = Wfiz + (long long)(32 + wv) * NKG * 512;
  const int rbase  = wv * NKG * 512;  // element offset of this wave's W_r tile

  for (int t = 0; t < T; ++t) {
    __syncthreads();  // A = [u_t ++ h_t] ready in LDS

    // Laundered zero offsets: formally loop-variant so the weight loads are
    // re-issued every step (streamed from L2 / re-read from LDS) instead of
    // being hoisted out of the T loop and spilled to scratch.
    int goff = 0;   // global (W_fiz) element offset
    int loff = 0;   // LDS (W_r / W_out) element offset
    asm volatile("" : "+s"(goff), "+s"(loff));

    // Fused gate GEMM: this wave computes f,i,z,r for its 16 hidden columns
    f32x8 aF = {}, aI = {}, aZ = {}, aR = {};
#pragma unroll
    for (int kci = 0; kci < NKG; ++kci) {
      bf16x16 a = load_a_frag(sA, lane, kci * 32);
      const int fo = (kci * 32 + lane) * 16;
      bf16x16 Bf = load_b_frag(pF + goff + fo);
      bf16x16 Bi = load_b_frag(pI + goff + fo);
      bf16x16 Bz = load_b_frag(pZ + goff + fo);
      bf16x16 Br = load_b_frag(&sWr[loff + rbase + fo]);
      aF = __builtin_amdgcn_wmma_f32_16x16x32_bf16(false, a, false, Bf, (short)0, aF, false, false);
      aI = __builtin_amdgcn_wmma_f32_16x16x32_bf16(false, a, false, Bi, (short)0, aI, false, false);
      aZ = __builtin_amdgcn_wmma_f32_16x16x32_bf16(false, a, false, Bz, (short)0, aZ, false, false);
      aR = __builtin_amdgcn_wmma_f32_16x16x32_bf16(false, a, false, Br, (short)0, aR, false, false);
    }

    // Elementwise recurrence (c stays in VGPRs)
    f32x8 h;
#pragma unroll
    for (int v = 0; v < 8; ++v) {
      float f  = sigm(aF[v] + bF);
      float ii = sigm(aI[v] + bI);
      float z  = sigm(aZ[v] + bZ);
      float r  = ftanh(aR[v] + bR);
      float cn = f * c[v] + ii * r;
      c[v] = cn;
      h[v] = z * ftanh(cn);
    }

    __syncthreads();  // everyone finished reading A(u_t, h_t)

    // h_{t+1} -> LDS (C layout maps lane->column, VGPR v->row)
#pragma unroll
    for (int v = 0; v < 8; ++v) {
      int m = v + 8 * rhalf;
      sA[m * LDA + D + colH] = (__bf16)h[v];
    }
    // stream next u tile
    if (t + 1 < T) {
      for (int i = tid; i < 16 * D; i += THREADS) {
        int m = i >> 6, n = i & 63;
        sA[m * LDA + n] = (__bf16)u[((long long)(b0 + m) * T + (t + 1)) * D + n];
      }
    }

    __syncthreads();  // h_{t+1} visible

    // Output projection y_t = h_{t+1} @ W_out + b_out (waves 0,1; wave-uniform branch)
    if (wv < 2) {
      f32x8 aY = {};
#pragma unroll
      for (int kci = 0; kci < NKO; ++kci) {
        bf16x16 a = load_a_frag(sA, lane, D + kci * 32);
        bf16x16 Bo = load_b_frag(&sWout[loff + ((wv * NKO + kci) * 32 + lane) * 16]);
        aY = __builtin_amdgcn_wmma_f32_16x16x32_bf16(false, a, false, Bo, (short)0, aY, false, false);
      }
#pragma unroll
      for (int v = 0; v < 8; ++v) {
        int m = v + 8 * rhalf;
        out[((long long)(b0 + m) * T + t) * O + wv * 16 + nloc] = aY[v] + bO;
      }
    }
  }
}

// ---------------------------------------------------------------------------
extern "C" void kernel_launch(void* const* d_in, const int* in_sizes, int n_in,
                              void* d_out, int out_size, void* d_ws, size_t ws_size,
                              hipStream_t stream) {
  (void)in_sizes; (void)n_in; (void)out_size; (void)ws_size;
  using namespace cfg;

  const float* u          = (const float*)d_in[0];
  const float* x0         = (const float*)d_in[1];
  const float* kernel_fiz = (const float*)d_in[2];
  const float* bias_fiz   = (const float*)d_in[3];
  const float* kernel_r   = (const float*)d_in[4];
  const float* bias_r     = (const float*)d_in[5];
  const float* W_out      = (const float*)d_in[6];
  const float* b_out      = (const float*)d_in[7];
  float* out = (float*)d_out;

  // Workspace: bf16 weights in wave32 B-fragment order (~656 KB total)
  __bf16* Wfiz_p = (__bf16*)d_ws;                       // 320*768
  __bf16* Wr_p   = Wfiz_p + (long long)KTOT * (3 * H);  // 320*256
  __bf16* Wout_p = Wr_p   + (long long)KTOT * H;        // 256*32

  auto packN = [](long long Ktot, long long Ntot) { return (Ntot / 16) * (Ktot / 32) * 512; };
  {
    long long n = packN(KTOT, 3 * H);
    pack_weights_kernel<<<(unsigned)((n + 255) / 256), 256, 0, stream>>>(kernel_fiz, Wfiz_p, KTOT, 3 * H);
  }
  {
    long long n = packN(KTOT, H);
    pack_weights_kernel<<<(unsigned)((n + 255) / 256), 256, 0, stream>>>(kernel_r, Wr_p, KTOT, H);
  }
  {
    long long n = packN(H, O);
    pack_weights_kernel<<<(unsigned)((n + 255) / 256), 256, 0, stream>>>(W_out, Wout_p, H, O);
  }

  // 8 persistent workgroups (one per 16-row batch tile), 16 wave32 each.
  lstm_scan_kernel<<<B / 16, THREADS, 0, stream>>>(
      u, x0, bias_fiz, bias_r, b_out, Wfiz_p, Wr_p, Wout_p, out);
}
